// WavetableSynth_84250078478345
// MI455X (gfx1250) — matrix operand order repaired and checked by reading, not compiled, not run on hardware
//
#include <hip/hip_runtime.h>
#include <hip/hip_bf16.h>
#include <math.h>

typedef __attribute__((ext_vector_type(16))) _Float16 v16h;
typedef __attribute__((ext_vector_type(16), aligned(16))) _Float16 v16h_a;  // 16B-aligned LDS view
typedef __attribute__((ext_vector_type(8)))  float    v8f;
typedef __attribute__((ext_vector_type(4)))  unsigned int u32x4;
typedef __attribute__((ext_vector_type(8)))  int      i32x8;
typedef __attribute__((ext_vector_type(4)))  int      i32x4;

#define D 128
#define LDSP 136     // padded LDS row stride in halves: 272B = 68 bank-words
                     // -> consecutive rows shift 4 banks; 16 rows tile all 64 banks
#define LDSPW 68     // padded LDS row stride in dwords
#define N_FRAMES 128
#define N_SAMPLES 32768
#define N_TABLES 8
#define TABLE_SIZE 512
#define BATCH 8
#define PI_F 3.14159265358979323846f

#if __has_builtin(__builtin_amdgcn_tensor_load_to_lds) && \
    __has_builtin(__builtin_amdgcn_s_wait_tensorcnt)
#define HAVE_TDM 1
#warning CDNA5 probe: TDM tensor_load_to_lds path ENABLED in this pass
#else
#define HAVE_TDM 0
#warning CDNA5 probe: TDM builtins unavailable in this pass - cooperative-copy fallback
#endif

__device__ __forceinline__ float lrelu(float v) { return v > 0.0f ? v : 0.2f * v; }

__device__ __forceinline__ v8f wmma_f16(v16h a, v16h b, v8f c) {
  // D = A(16x32,f16) * B(32x16,f16) + C(16x16,f32)
  return __builtin_amdgcn_wmma_f32_16x16x32_f16(false, a, false, b, (short)0, c, false, false);
}

// Gather a 16x32 f16 A-fragment from a bank-padded row-major [16][LDSP] LDS tile.
// 16-bit A 16x32: lane = (half<<4)|m ; VGPR j: K = 2j + 16*half (contiguous,
// matching loadB); pairs (K,K+1) per VGPR.
__device__ __forceinline__ v16h gatherA(const _Float16* h, int lane, int kstep) {
  const int m = lane & 15, g = lane >> 4;
  const _Float16* row = h + m * LDSP + kstep * 32 + 16 * g;
  v16h a;
#pragma unroll
  for (int j = 0; j < 8; ++j) {
    const int kb = 2 * j;
    a[2 * j]     = row[kb];
    a[2 * j + 1] = row[kb + 1];
  }
  return a;
}

// Load a 32x16 f16 B-fragment from a bank-padded N-major [rows][LDSP] LDS tile.
// lanes 0-15: K=k0..k0+15, lanes 16-31: K=k0+16..k0+31; N = n0 + (lane&15).
// 16 contiguous halves -> two ds_load_b128; padded stride -> ~conflict-free.
__device__ __forceinline__ v16h loadB(const _Float16* wT, int lane, int n0, int k0) {
  const int n = n0 + (lane & 15);
  const int k = k0 + 16 * (lane >> 4);
  const v16h_a t = *(const v16h_a*)(wT + n * LDSP + k);
  return __builtin_bit_cast(v16h, t);
}

#if HAVE_TDM
// Tensor Data Mover: DMA a [rows][128] f16 tile (contiguous in global) into
// LDS with hardware row padding (pad_interval=5 -> 64 DW rows, pad_amount=3 ->
// +4 DW) producing the bank-staggered [rows][LDSP] layout directly.
// D# per CDNA5 ISA.
__device__ __forceinline__ void tdm_load_tile_f16(const _Float16* gsrc,
                                                  unsigned lds_off, int rows) {
  const unsigned long long ga = (unsigned long long)(size_t)gsrc;
  u32x4 g0;
  g0[0] = 1u;                                      // count=1, user mode
  g0[1] = lds_off;                                 // lds_addr (bytes)
  g0[2] = (unsigned)(ga & 0xffffffffu);            // global_addr[31:0]
  g0[3] = (unsigned)((ga >> 32) & 0x01ffffffu) | (2u << 30);  // addr[56:32], type=2
  i32x8 g1;
  g1[0] = (int)((1u << 16)      // data_size = 1 (2 bytes)
              | (1u << 20)      // pad_enable
              | (5u << 22)      // pad_interval: 64 DWORDs (one 128-half row)
              | (3u << 25));    // pad_amount: 4 DWORDs (8 halves)
  g1[1] = (int)(128u << 16);                       // tensor_dim0 = 128 (lo16)
  g1[2] = (int)(((unsigned)rows) << 16);           // dim0 hi=0 | tensor_dim1 lo16
  g1[3] = (int)(128u << 16);                       // dim1 hi=0 | tile_dim0 = 128
  g1[4] = rows;                                    // tile_dim1 = rows, tile_dim2 = 0
  g1[5] = 128;                                     // tensor_dim0_stride = 128
  g1[6] = 0;
  g1[7] = 0;
  const i32x4 z4 = {0, 0, 0, 0};
#if __clang_major__ >= 23
  const i32x8 z8 = {0, 0, 0, 0, 0, 0, 0, 0};
  __builtin_amdgcn_tensor_load_to_lds(g0, g1, z4, z4, z8, 0);
#else
  __builtin_amdgcn_tensor_load_to_lds(g0, g1, z4, z4, 0);
#endif
}
#endif

// Issue staging of a [rows][128] f16 tile into bank-padded LDS (TDM from wave
// 0, or cooperative coalesced copy fallback). Pair with stage_wait+barrier.
__device__ __forceinline__ void stage_issue(_Float16* dst, const _Float16* src,
                                            int rows, int tid) {
#if HAVE_TDM
  if ((tid >> 5) == 0) tdm_load_tile_f16(src, (unsigned)(size_t)dst, rows);
#else
  unsigned int* d = (unsigned int*)dst;
  const unsigned int* s = (const unsigned int*)src;
  for (int i = tid; i < rows * 64; i += 128) {
    const int r = i >> 6, cd = i & 63;
    d[r * LDSPW + cd] = s[i];
  }
#endif
}

__device__ __forceinline__ void stage_wait(int tid) {
#if HAVE_TDM
  if ((tid >> 5) == 0) __builtin_amdgcn_s_wait_tensorcnt(0);
#endif
  (void)tid;
}

// ---------------------------------------------------------------------------
// Kernel 1: tiny setup. tc-MLP -> softmax max/argmax -> values + selected
// normalized table; env-MLP -> |.|; xw1 = x@sk_w1 + sk_b1. Also one-time
// transpose+convert of all WMMA weights to f16 N-major tiles. One block.
// ---------------------------------------------------------------------------
__global__ __launch_bounds__(256) void synth_setup_kernel(
    const float* __restrict__ x, const float* __restrict__ wt,
    const float* __restrict__ tw1, const float* __restrict__ tb1,
    const float* __restrict__ tw2, const float* __restrict__ tb2,
    const float* __restrict__ tw3, const float* __restrict__ tb3,
    const float* __restrict__ ew1, const float* __restrict__ eb1,
    const float* __restrict__ ew2, const float* __restrict__ eb2,
    const float* __restrict__ ew3, const float* __restrict__ eb3,
    const float* __restrict__ skw1, const float* __restrict__ skb1,
    const float* __restrict__ pw2, const float* __restrict__ pw3,
    const float* __restrict__ skw2, const float* __restrict__ skw3,
    float* __restrict__ selW, float* __restrict__ valW,
    float* __restrict__ envW, float* __restrict__ xw1W,
    _Float16* __restrict__ pw2T, _Float16* __restrict__ pw3T,
    _Float16* __restrict__ skw1T, _Float16* __restrict__ w2T,
    _Float16* __restrict__ w3T) {
  __shared__ float h1[D], h2[D], lg[N_TABLES], wmax[N_TABLES];
  __shared__ int selIdx;
  const int tid = threadIdx.x;

  // ---- one-time weight transpose + f16 convert (N-major [n][k]) ----
  for (int i = tid; i < D * D; i += blockDim.x) {
    const int k = i >> 7, n = i & 127;
    pw2T[n * D + k]  = (_Float16)pw2[i];
    pw3T[n * D + k]  = (_Float16)pw3[i];
    skw1T[n * D + k] = (_Float16)skw1[i];
    w2T[n * D + k]   = (_Float16)skw2[i];
  }
  for (int i = tid; i < D * TABLE_SIZE; i += blockDim.x) {
    const int k = i >> 9, n = i & 511;   // src skw3[k][n], n<512
    w3T[n * D + k] = (_Float16)skw3[i];
  }

  if (tid < N_TABLES) {
    float m = -3.0e38f;
    for (int k = 0; k < TABLE_SIZE; ++k) m = fmaxf(m, wt[tid * TABLE_SIZE + k]);
    wmax[tid] = m;
  }
  __syncthreads();

  for (int b = 0; b < BATCH; ++b) {
    // ---- table-choice MLP ----
    if (tid < D) {
      float a = tb1[tid];
      for (int i = 0; i < D; ++i) a += x[b * D + i] * tw1[i * D + tid];
      h1[tid] = lrelu(a);
    }
    __syncthreads();
    if (tid < D) {
      float a = tb2[tid];
      for (int k = 0; k < D; ++k) a += h1[k] * tw2[k * D + tid];
      h2[tid] = lrelu(a);
    }
    __syncthreads();
    if (tid < N_TABLES) {
      float a = tb3[tid];
      for (int k = 0; k < D; ++k) a += h2[k] * tw3[k * N_TABLES + tid];
      lg[tid] = a;
    }
    __syncthreads();
    if (tid == 0) {
      float m = lg[0]; int idx = 0;
      for (int k = 1; k < N_TABLES; ++k) if (lg[k] > m) { m = lg[k]; idx = k; }
      float s = 0.0f;
      for (int k = 0; k < N_TABLES; ++k) s += __expf(lg[k] - m);
      valW[b] = 1.0f / s;  // max softmax probability
      selIdx = idx;
    }
    __syncthreads();
    {
      const int idx = selIdx;
      const float inv = 1.0f / (wmax[idx] + 1e-8f);
      for (int k = tid; k < TABLE_SIZE; k += blockDim.x)
        selW[b * TABLE_SIZE + k] = wt[idx * TABLE_SIZE + k] * inv;
    }
    // ---- envelope MLP ----
    if (tid < D) {
      float a = eb1[tid];
      for (int i = 0; i < D; ++i) a += x[b * D + i] * ew1[i * D + tid];
      h1[tid] = lrelu(a);
    }
    __syncthreads();
    if (tid < D) {
      float a = eb2[tid];
      for (int k = 0; k < D; ++k) a += h1[k] * ew2[k * D + tid];
      h2[tid] = lrelu(a);
    }
    __syncthreads();
    if (tid < N_FRAMES) {
      float a = eb3[tid];
      for (int k = 0; k < D; ++k) a += h2[k] * ew3[k * N_FRAMES + tid];
      envW[b * N_FRAMES + tid] = fabsf(a);
    }
    // ---- xw1 = x @ sk_w1 + sk_b1 ----
    if (tid < D) {
      float a = skb1[tid];
      for (int i = 0; i < D; ++i) a += x[b * D + i] * skw1[i * D + tid];
      xw1W[b * D + tid] = a;
    }
    __syncthreads();
  }
}

// ---------------------------------------------------------------------------
// Kernel 2 (WMMA + TDM): batch-independent token path.
// pe(t) (K=33, scalar) -> [@pos_w2, @pos_w3, @sk_w1] via WMMA -> P1 f16.
// Weight tiles arrive by tensor DMA (bank-padded) overlapped with VALU work.
// ---------------------------------------------------------------------------
__global__ __launch_bounds__(128) void synth_pos_kernel(
    const float* __restrict__ pw1, const float* __restrict__ pb1,
    const float* __restrict__ pb2, const float* __restrict__ pb3,
    const _Float16* __restrict__ pw2T, const _Float16* __restrict__ pw3T,
    const _Float16* __restrict__ skw1T, _Float16* __restrict__ P1) {
  __shared__ __align__(32) _Float16 wBufT[D * LDSP];   // weight tile, N-major padded
  __shared__ __align__(32) _Float16 hBuf[64 * LDSP];   // per-wave 16xLDSP activations
  __shared__ float peS[64 * 33];

  const int tid = threadIdx.x;
  const int lane = tid & 31;
  const int wave = tid >> 5;
  const int tBlock = blockIdx.x * 64;
  const int g = lane >> 4;
  const int nl = lane & 15;

  // ---- kick off W2 DMA, then build positional encoding under it ----
  stage_issue(wBufT, pw2T, D, tid);
  for (int i = tid; i < 64 * 33; i += 128) {
    const int t = i / 33, j = i - t * 33;
    const float p = -1.0f + 2.0f * (float)(tBlock + t) * (1.0f / 32767.0f);
    float v;
    if (j == 0)        v = p;
    else if (j <= 16)  v = __sinf(p * exp2f((float)(j - 1))  * PI_F);
    else               v = __cosf(p * exp2f((float)(j - 17)) * PI_F);
    peS[i] = v;
  }
  __syncthreads();

  // ---- layer 1 (K=33): scalar, one channel per thread ----
  {
    const int c = tid;
    float wcol[33];
#pragma unroll
    for (int i = 0; i < 33; ++i) wcol[i] = pw1[i * D + c];
    const float bias = pb1[c];
    for (int t = 0; t < 64; ++t) {
      float a = bias;
#pragma unroll
      for (int i = 0; i < 33; ++i) a += peS[t * 33 + i] * wcol[i];
      hBuf[t * LDSP + c] = (_Float16)lrelu(a);
    }
  }
  stage_wait(tid);
  __syncthreads();

  const _Float16* hb = hBuf + wave * 16 * LDSP;
  v16h a[4];

  // ---- layer 2: h2 = lrelu(h1 @ pos_w2 + b2) ----
#pragma unroll
  for (int ks = 0; ks < 4; ++ks) a[ks] = gatherA(hb, lane, ks);
#pragma unroll
  for (int nt = 0; nt < 8; ++nt) {
    const float bias = pb2[nt * 16 + nl];
    v8f acc;
#pragma unroll
    for (int j = 0; j < 8; ++j) acc[j] = bias;
#pragma unroll
    for (int ks = 0; ks < 4; ++ks)
      acc = wmma_f16(a[ks], loadB(wBufT, lane, nt * 16, ks * 32), acc);
#pragma unroll
    for (int j = 0; j < 8; ++j)
      hBuf[(wave * 16 + j + 8 * g) * LDSP + nt * 16 + nl] = (_Float16)lrelu(acc[j]);
  }
  __syncthreads();

  // ---- layer 3: pos = h2 @ pos_w3 + b3 (no activation) ----
  stage_issue(wBufT, pw3T, D, tid);
#pragma unroll
  for (int ks = 0; ks < 4; ++ks) a[ks] = gatherA(hb, lane, ks);
  stage_wait(tid);
  __syncthreads();
#pragma unroll
  for (int nt = 0; nt < 8; ++nt) {
    const float bias = pb3[nt * 16 + nl];
    v8f acc;
#pragma unroll
    for (int j = 0; j < 8; ++j) acc[j] = bias;
#pragma unroll
    for (int ks = 0; ks < 4; ++ks)
      acc = wmma_f16(a[ks], loadB(wBufT, lane, nt * 16, ks * 32), acc);
#pragma unroll
    for (int j = 0; j < 8; ++j)
      hBuf[(wave * 16 + j + 8 * g) * LDSP + nt * 16 + nl] = (_Float16)acc[j];
  }
  __syncthreads();

  // ---- layer 4: P1 = pos @ sk_w1 (bias folded into xw1 later) ----
  stage_issue(wBufT, skw1T, D, tid);
#pragma unroll
  for (int ks = 0; ks < 4; ++ks) a[ks] = gatherA(hb, lane, ks);
  stage_wait(tid);
  __syncthreads();
#pragma unroll
  for (int nt = 0; nt < 8; ++nt) {
    v8f acc;
#pragma unroll
    for (int j = 0; j < 8; ++j) acc[j] = 0.0f;
#pragma unroll
    for (int ks = 0; ks < 4; ++ks)
      acc = wmma_f16(a[ks], loadB(wBufT, lane, nt * 16, ks * 32), acc);
#pragma unroll
    for (int j = 0; j < 8; ++j)
      hBuf[(wave * 16 + j + 8 * g) * LDSP + nt * 16 + nl] = (_Float16)acc[j];
  }
  __syncthreads();

  // ---- coalesced packed store of the 64x128 f16 tile (de-pad rows) ----
  {
    const unsigned int* hu = (const unsigned int*)hBuf;
    unsigned int* Pu = (unsigned int*)(P1 + (size_t)tBlock * D);
#pragma unroll 4
    for (int i = tid; i < 64 * 64; i += 128) {
      const int t = i >> 6, cd = i & 63;
      Pu[i] = hu[t * LDSPW + cd];
    }
  }
}

// ---------------------------------------------------------------------------
// Kernel 3: fused main path. 4 waves / block, 16 tokens per wave.
// h1 = lrelu(P1 + xw1[b]); H2 = lrelu(h1@W2 + b2) via WMMA;
// logits = H2@W3 + b3 via WMMA in 8 double-buffered 64-col chunks (TDM DMA
// overlapped with compute) + online softmax·sel dot;
// out = (dot/sum) * env_interp * values[b].
// ---------------------------------------------------------------------------
__global__ __launch_bounds__(128) void synth_main_kernel(
    const _Float16* __restrict__ P1, const float* __restrict__ selW,
    const float* __restrict__ valW, const float* __restrict__ envW,
    const float* __restrict__ xw1W,
    const _Float16* __restrict__ w2T, const float* __restrict__ b2,
    const _Float16* __restrict__ w3T, const float* __restrict__ b3,
    float* __restrict__ out) {
  __shared__ __align__(32) _Float16 wBufT[2 * 64 * LDSP];  // 2 chunk buffers / W2 tile
  __shared__ __align__(32) _Float16 hBuf[64 * LDSP];       // per-wave activations
  __shared__ float selS[TABLE_SIZE];
  __shared__ float b3S[TABLE_SIZE];

  const int tid = threadIdx.x;
  const int lane = tid & 31;
  const int wave = tid >> 5;
  const int b = blockIdx.y;
  const int tBlock = blockIdx.x * 64;
  const int t0 = tBlock + wave * 16;
  const int g = lane >> 4;
  const int nl = lane & 15;

  // ---- kick off W2 DMA; stage sel, b3, h1 under it ----
  stage_issue(wBufT, w2T, D, tid);
  for (int i = tid; i < TABLE_SIZE; i += 128) {
    selS[i] = selW[b * TABLE_SIZE + i];
    b3S[i]  = b3[i];
  }
#pragma unroll 4
  for (int i = tid; i < 64 * D; i += 128) {
    const int t = i >> 7, c = i & 127;
    const float h = (float)P1[(tBlock + t) * D + c] + xw1W[b * D + c];
    hBuf[t * LDSP + c] = (_Float16)lrelu(h);
  }
  stage_wait(tid);
  __syncthreads();

  const _Float16* hb = hBuf + wave * 16 * LDSP;
  v16h a1[4];
#pragma unroll
  for (int ks = 0; ks < 4; ++ks) a1[ks] = gatherA(hb, lane, ks);

  // ---- layer 2 GEMM (16x128 @ 128x128) ----
#pragma unroll
  for (int nt = 0; nt < 8; ++nt) {
    const float bias = b2[nt * 16 + nl];
    v8f acc;
#pragma unroll
    for (int j = 0; j < 8; ++j) acc[j] = bias;
#pragma unroll
    for (int ks = 0; ks < 4; ++ks)
      acc = wmma_f16(a1[ks], loadB(wBufT, lane, nt * 16, ks * 32), acc);
#pragma unroll
    for (int j = 0; j < 8; ++j)
      hBuf[(wave * 16 + j + 8 * g) * LDSP + nt * 16 + nl] = (_Float16)lrelu(acc[j]);
  }
  __syncthreads();

  v16h a2[4];
#pragma unroll
  for (int ks = 0; ks < 4; ++ks) a2[ks] = gatherA(hb, lane, ks);

  float rowM[8], rowS[8], rowD[8];
#pragma unroll
  for (int j = 0; j < 8; ++j) { rowM[j] = -3.0e38f; rowS[j] = 0.0f; rowD[j] = 0.0f; }

  // ---- prologue: DMA chunk 0 into buffer 0 ----
  stage_issue(wBufT, w3T, 64, tid);
  stage_wait(tid);
  __syncthreads();

  // ---- layer 3 (16x128 @ 128x512) in 8 double-buffered 64-col chunks ----
  for (int ch = 0; ch < 8; ++ch) {
    _Float16* cur = wBufT + (ch & 1) * 64 * LDSP;
    _Float16* nxt = wBufT + ((ch + 1) & 1) * 64 * LDSP;
    if (ch < 7) stage_issue(nxt, w3T + (size_t)(ch + 1) * 64 * D, 64, tid);

    v8f acc[4];
#pragma unroll
    for (int nt = 0; nt < 4; ++nt) {
      const float bias = b3S[ch * 64 + nt * 16 + nl];
#pragma unroll
      for (int j = 0; j < 8; ++j) acc[nt][j] = bias;
#pragma unroll
      for (int ks = 0; ks < 4; ++ks)
        acc[nt] = wmma_f16(a2[ks], loadB(cur, lane, nt * 16, ks * 32), acc[nt]);
    }

    float cm[8];
#pragma unroll
    for (int j = 0; j < 8; ++j) {
      cm[j] = acc[0][j];
#pragma unroll
      for (int nt = 1; nt < 4; ++nt) cm[j] = fmaxf(cm[j], acc[nt][j]);
    }
#pragma unroll
    for (int m = 1; m < 16; m <<= 1)
#pragma unroll
      for (int j = 0; j < 8; ++j) cm[j] = fmaxf(cm[j], __shfl_xor(cm[j], m, 16));

#pragma unroll
    for (int j = 0; j < 8; ++j) {
      const float nM = fmaxf(rowM[j], cm[j]);
      const float sc = __expf(rowM[j] - nM);
      rowS[j] *= sc; rowD[j] *= sc; rowM[j] = nM;
    }
    float ps[8], pd[8];
#pragma unroll
    for (int j = 0; j < 8; ++j) { ps[j] = 0.0f; pd[j] = 0.0f; }
#pragma unroll
    for (int nt = 0; nt < 4; ++nt) {
      const float sv = selS[ch * 64 + nt * 16 + nl];
#pragma unroll
      for (int j = 0; j < 8; ++j) {
        const float e = __expf(acc[nt][j] - rowM[j]);
        ps[j] += e; pd[j] += e * sv;
      }
    }
#pragma unroll
    for (int m = 1; m < 16; m <<= 1)
#pragma unroll
      for (int j = 0; j < 8; ++j) {
        ps[j] += __shfl_xor(ps[j], m, 16);
        pd[j] += __shfl_xor(pd[j], m, 16);
      }
#pragma unroll
    for (int j = 0; j < 8; ++j) { rowS[j] += ps[j]; rowD[j] += pd[j]; }

    __syncthreads();              // all waves done reading `cur`
    if (ch < 7) stage_wait(tid);  // `nxt` DMA complete
    __syncthreads();
  }

  // ---- env interpolation * values, store 16 samples / wave ----
  if (nl == 0) {
    const float vb = valW[b];
#pragma unroll
    for (int j = 0; j < 8; ++j) {
      const int t = t0 + j + 8 * g;
      const float coord = ((float)t + 0.5f) * (1.0f / 256.0f) - 0.5f;
      const float fl = floorf(coord);
      const float fr = coord - fl;
      int i0 = (int)fl;
      const int i0c = i0 < 0 ? 0 : (i0 > N_FRAMES - 1 ? N_FRAMES - 1 : i0);
      int i1 = i0 + 1;
      const int i1c = i1 < 0 ? 0 : (i1 > N_FRAMES - 1 ? N_FRAMES - 1 : i1);
      const float e = envW[b * N_FRAMES + i0c] * (1.0f - fr) + envW[b * N_FRAMES + i1c] * fr;
      out[b * N_SAMPLES + t] = (rowD[j] / rowS[j]) * e * vb;
    }
  }
}

// ---------------------------------------------------------------------------
extern "C" void kernel_launch(void* const* d_in, const int* in_sizes, int n_in,
                              void* d_out, int out_size, void* d_ws, size_t ws_size,
                              hipStream_t stream) {
  (void)in_sizes; (void)n_in; (void)out_size; (void)ws_size;
  const float* x    = (const float*)d_in[0];
  const float* wtab = (const float*)d_in[1];
  const float* tw1 = (const float*)d_in[2];  const float* tb1 = (const float*)d_in[3];
  const float* tw2 = (const float*)d_in[4];  const float* tb2 = (const float*)d_in[5];
  const float* tw3 = (const float*)d_in[6];  const float* tb3 = (const float*)d_in[7];
  const float* ew1 = (const float*)d_in[8];  const float* eb1 = (const float*)d_in[9];
  const float* ew2 = (const float*)d_in[10]; const float* eb2 = (const float*)d_in[11];
  const float* ew3 = (const float*)d_in[12]; const float* eb3 = (const float*)d_in[13];
  const float* pw1 = (const float*)d_in[14]; const float* pb1 = (const float*)d_in[15];
  const float* pw2 = (const float*)d_in[16]; const float* pb2 = (const float*)d_in[17];
  const float* pw3 = (const float*)d_in[18]; const float* pb3 = (const float*)d_in[19];
  const float* skw1 = (const float*)d_in[20]; const float* skb1 = (const float*)d_in[21];
  const float* skw2 = (const float*)d_in[22]; const float* skb2 = (const float*)d_in[23];
  const float* skw3 = (const float*)d_in[24]; const float* skb3 = (const float*)d_in[25];
  float* out = (float*)d_out;

  // workspace layout (64B aligned sections)
  char* ws = (char*)d_ws;
  float*     selW  = (float*)(ws);                  // 8*512 f32   @ 0
  float*     valW  = (float*)(ws + 16384);          // 8 f32       @ 16384
  float*     envW  = (float*)(ws + 16448);          // 8*128 f32   @ 16448
  float*     xw1W  = (float*)(ws + 20544);          // 8*128 f32   @ 20544
  _Float16*  P1    = (_Float16*)(ws + 24640);       // 32768*128 f16 (8 MiB)
  _Float16*  pw2T  = (_Float16*)(ws + 8413248);     // 128*128 f16
  _Float16*  pw3T  = (_Float16*)(ws + 8446016);     // 128*128 f16
  _Float16*  skw1T = (_Float16*)(ws + 8478784);     // 128*128 f16
  _Float16*  w2T   = (_Float16*)(ws + 8511552);     // 128*128 f16
  _Float16*  w3T   = (_Float16*)(ws + 8544320);     // 512*128 f16 (N-major)

  synth_setup_kernel<<<1, 256, 0, stream>>>(
      x, wtab, tw1, tb1, tw2, tb2, tw3, tb3,
      ew1, eb1, ew2, eb2, ew3, eb3, skw1, skb1,
      pw2, pw3, skw2, skw3,
      selW, valW, envW, xw1W, pw2T, pw3T, skw1T, w2T, w3T);

  synth_pos_kernel<<<N_SAMPLES / 64, 128, 0, stream>>>(
      pw1, pb1, pb2, pb3, pw2T, pw3T, skw1T, P1);

  synth_main_kernel<<<dim3(N_SAMPLES / 64, BATCH), 128, 0, stream>>>(
      P1, selW, valW, envW, xw1W, w2T, skb2, w3T, skb3, out);
}
